// STGCN_11682311045804
// MI455X (gfx1250) — matrix operand order repaired and checked by reading, not compiled, not run on hardware
//
#include <hip/hip_runtime.h>

typedef __attribute__((ext_vector_type(16))) __bf16 v16bf;
typedef __attribute__((ext_vector_type(8)))  __bf16 v8bf;
typedef __attribute__((ext_vector_type(8)))  float  v8f;

#define BN_EPS 1e-5f

// ---------------------------------------------------------------------------
// data_bn: x (32,3,300,25,2) f32 -> actA [((n*2+m)*300+t)*25+v]*3+c  bf16
// BN channel index = v*3 + c (reference transposes to (n*m, v*c, t))
// ---------------------------------------------------------------------------
__global__ void data_bn_kernel(const float* __restrict__ x,
                               const float* __restrict__ g,
                               const float* __restrict__ bt,
                               const float* __restrict__ mu,
                               const float* __restrict__ va,
                               __bf16* __restrict__ out, long total)
{
  long i = (long)blockIdx.x * blockDim.x + threadIdx.x;
  if (i >= total) return;
  int c = (int)(i % 3);
  long r = i / 3;
  int v = (int)(r % 25); r /= 25;
  int t = (int)(r % 300); r /= 300;
  int m = (int)(r % 2);
  int n = (int)(r / 2);
  int ch = v * 3 + c;
  float sc  = g[ch] * rsqrtf(va[ch] + BN_EPS);
  float val = x[((((long)n * 3 + c) * 300 + t) * 25 + v) * 2 + m];
  val = (val - mu[ch]) * sc + bt[ch];
  out[i] = (__bf16)val;
}

// ---------------------------------------------------------------------------
// Small prep kernels (per block)
// ---------------------------------------------------------------------------
__global__ void prep_ai_kernel(const float* __restrict__ A,
                               const float* __restrict__ imp,
                               float* __restrict__ Ai)
{
  int i = blockIdx.x * blockDim.x + threadIdx.x;
  if (i < 1875) Ai[i] = A[i] * imp[i];
}

// bias_eff[c][w] = sum_k gcn_b[k*cout+c] * sum_v Ai[k,v,w]
__global__ void prep_bias_eff_kernel(const float* __restrict__ gb,
                                     const float* __restrict__ Ai,
                                     float* __restrict__ be, int cout)
{
  int i = blockIdx.x * blockDim.x + threadIdx.x;
  if (i >= cout * 25) return;
  int c = i / 25, w = i - c * 25;
  float s = 0.f;
  for (int k = 0; k < 3; ++k) {
    float cs = 0.f;
    for (int v = 0; v < 25; ++v) cs += Ai[k * 625 + v * 25 + w];
    s += gb[k * cout + c] * cs;
  }
  be[i] = s;
}

// fused gcn weight: WG[c][r], r = k*cin+ic  (zero-padded to Kp)
__global__ void prep_gcnw_kernel(const float* __restrict__ w,
                                 __bf16* __restrict__ o,
                                 int cout, int cin, int Kp)
{
  int i = blockIdx.x * blockDim.x + threadIdx.x;
  if (i >= cout * Kp) return;
  int c = i / Kp, r = i - c * Kp;
  float val = 0.f;
  if (r < 3 * cin) {
    int k = r / cin, ic = r - k * cin;
    val = w[(long)(k * cout + c) * cin + ic];
  }
  o[i] = (__bf16)val;
}

// tcn weight: WT[oc][dt*cout+ic] <- tcn_w[oc][ic][dt]
__global__ void prep_tcnw_kernel(const float* __restrict__ w,
                                 __bf16* __restrict__ o, int cout)
{
  int K = cout * 9;
  int i = blockIdx.x * blockDim.x + threadIdx.x;
  if (i >= cout * K) return;
  int oc = i / K, r = i - oc * K;
  int dt = r / cout, ic = r - dt * cout;
  o[i] = (__bf16)w[(long)(oc * cout + ic) * 9 + dt];
}

__global__ void prep_resw_kernel(const float* __restrict__ w,
                                 __bf16* __restrict__ o, int n)
{
  int i = blockIdx.x * blockDim.x + threadIdx.x;
  if (i < n) o[i] = (__bf16)w[i];
}

// ---------------------------------------------------------------------------
// Graph aggregation (commuted before 1x1 conv):
//   XA[((nt*25)+w)*Kp + (k*cin+ic)] = sum_v Xin[(nt*25+v)*cin + ic] * Ai[k,v,w]
// ---------------------------------------------------------------------------
__global__ void agg_kernel(const __bf16* __restrict__ Xin,
                           const float* __restrict__ Ai,
                           __bf16* __restrict__ XA,
                           long NT, int cin, int K3, int Kp)
{
  long i = (long)blockIdx.x * blockDim.x + threadIdx.x;
  long total = NT * 25 * (long)K3;
  if (i >= total) return;
  int r = (int)(i % K3);
  long p = i / K3;             // output pixel = nt*25 + w
  int w  = (int)(p % 25);
  long nt = p / 25;
  int k = r / cin, ic = r - k * cin;
  const __bf16* xrow = Xin + nt * 25 * (long)cin + ic;
  const float*  arow = Ai + k * 625 + w;
  float acc = 0.f;
#pragma unroll
  for (int v = 0; v < 25; ++v)
    acc += (float)xrow[(long)v * cin] * arow[v * 25];
  XA[p * Kp + r] = (__bf16)acc;
}

// ---------------------------------------------------------------------------
// WMMA GEMM: D[M x N] = W[M x Kp] * B[Kp x N]  (bf16 operands, f32 accum)
//   MODE 0: GCN  (B = XA, K-contiguous per pixel)    epilogue: +biasEff, bn1, relu
//   MODE 1: TCN  (implicit temporal im2col from H)   epilogue: +b, bn2, +res, relu
//   MODE 2: RES  (strided 1x1 conv from block input) epilogue: +b, res_bn
// Block: 256 threads = 8 waves as 2(M) x 4(N); wave tile 16M x 64N.
// All 4 B fragments are issued before the 4 WMMAs so the scheduler can clause
// the 8 b128 loads and retire WMMAs against staggered loadcnt waits.
// ---------------------------------------------------------------------------
struct GemmArgs {
  const __bf16* W;
  const __bf16* X;
  __bf16* Out;                 // [n][M] bf16
  const float* biasEff;        // MODE 0: [M][25]
  const float* cb;             // MODE 1/2: per-channel bias
  const float* bn_g; const float* bn_b; const float* bn_m; const float* bn_v;
  const __bf16* Res;           // MODE 1 residual source [n][M]
  float* OutF;                 // MODE 1 final f32 NCHW output (or null)
  int M, Nn, Kp;
  int Tin, Tout, V, stride, cx, resMode;
};

template<int MODE>
__global__ __launch_bounds__(256) void gemm_kernel(GemmArgs a)
{
  const int lane = threadIdx.x & 31;
  const int wave = threadIdx.x >> 5;
  const int wm = wave & 1, wn = wave >> 1;
  const int mBase = blockIdx.y * 32 + wm * 16;
  const int nBase = blockIdx.x * 256 + wn * 64;
  const int half = lane >> 4;       // lane group 0/1
  const int l16  = lane & 15;

  v8f acc[4];
  for (int s = 0; s < 4; ++s)
    for (int e = 0; e < 8; ++e) acc[s][e] = 0.f;

  const __bf16* Wrow = a.W + (long)(mBase + l16) * a.Kp;

  int pn[4], pnb[4], pto[4], pv[4];
  for (int s = 0; s < 4; ++s) {
    int n = nBase + s * 16 + l16;
    if (n >= a.Nn) n = a.Nn - 1;   // clamp: valid reads, store guarded
    pn[s] = n;
    if (MODE != 0) {
      int tv = a.Tout * a.V;
      pnb[s] = n / tv;
      int rem = n - pnb[s] * tv;
      pto[s] = rem / a.V;
      pv[s]  = rem - pto[s] * a.V;
    }
  }

  for (int k0 = 0; k0 < a.Kp; k0 += 32) {
    // A fragment: lanes 0-15: K k0..k0+15, lanes 16-31: +16 (matches B)
    v16bf afrag;
    {
      const __bf16* ap = Wrow + k0 + half * 16;
      afrag = *(const v16bf*)(ap);
    }
    __builtin_prefetch((const void*)(Wrow + k0 + 64), 0, 1);

    const int kB = k0 + half * 16;  // B layout: lanes 0-15 K k0..k0+15, 16-31 +16
    v16bf bf[4];
#pragma unroll
    for (int s = 0; s < 4; ++s) {
      if (MODE == 0) {
        bf[s] = *(const v16bf*)(a.X + (long)pn[s] * a.Kp + kB);
      } else if (MODE == 1) {
        int dt = kB / a.cx;           // constant within the 16-chunk (cx % 32 == 0)
        int ic = kB - dt * a.cx;
        int ti = pto[s] * a.stride + dt - 4;
        if (ti >= 0 && ti < a.Tin) {
          long pix = ((long)pnb[s] * a.Tin + ti) * a.V + pv[s];
          bf[s] = *(const v16bf*)(a.X + pix * a.cx + ic);
        } else {
#pragma unroll
          for (int i2 = 0; i2 < 16; ++i2) bf[s][i2] = (__bf16)0.f;
        }
      } else { // MODE 2: strided 1x1
        long pix = ((long)pnb[s] * a.Tin + pto[s] * a.stride) * a.V + pv[s];
        bf[s] = *(const v16bf*)(a.X + pix * a.cx + kB);
      }
    }
#pragma unroll
    for (int s = 0; s < 4; ++s) {
      acc[s] = __builtin_amdgcn_wmma_f32_16x16x32_bf16(
          false, afrag, false, bf[s], (short)0, acc[s], false, false);
    }
  }

  // Hoisted per-row epilogue params: t = acc*bnsc[e] + off[e] (+ graph bias)
  // D layout: elem e, lane l -> m = mBase + e + 8*(l>=16), n col = l%16
  float bnsc[8], off[8];
#pragma unroll
  for (int e = 0; e < 8; ++e) {
    int m = mBase + e + half * 8;
    float sc = a.bn_g[m] * rsqrtf(a.bn_v[m] + BN_EPS);
    bnsc[e] = sc;
    float o = a.bn_b[m] - a.bn_m[m] * sc;
    if (MODE != 0) o += a.cb[m] * sc;     // per-channel conv bias through BN scale
    off[e] = o;
  }

  for (int s = 0; s < 4; ++s) {
    int n = nBase + s * 16 + l16;
    if (n >= a.Nn) continue;
    float gb = 0.f;
    // note: for MODE 0, pixel's joint index v = n % V selects graph bias column
#pragma unroll
    for (int e = 0; e < 8; ++e) {
      int m = mBase + e + half * 8;
      float t = acc[s][e];
      if (MODE == 0) t += a.biasEff[m * a.V + (n % a.V)];
      t = t * bnsc[e] + off[e];
      if (MODE == 1) {
        if (a.resMode != 0) t += (float)a.Res[(long)n * a.M + m];
        t = fmaxf(t, 0.f);
      } else if (MODE == 0) {
        t = fmaxf(t, 0.f);
      }
      a.Out[(long)n * a.M + m] = (__bf16)t;
      if (MODE == 1 && a.OutF) {
        a.OutF[(((long)pnb[s] * a.M + m) * a.Tout + pto[s]) * a.V + pv[s]] = t;
      }
    }
    (void)gb;
  }
}

// ---------------------------------------------------------------------------
// Host
// ---------------------------------------------------------------------------
static inline int cdiv_i(int x, int y) { return (x + y - 1) / y; }

extern "C" void kernel_launch(void* const* d_in, const int* in_sizes, int n_in,
                              void* d_out, int out_size, void* d_ws, size_t ws_size,
                              hipStream_t stream)
{
  (void)in_sizes; (void)n_in; (void)out_size; (void)ws_size;

  // ---- inputs: depth-first, setup_inputs() insertion order ----
  int ix = 0;
  const float* X    = (const float*)d_in[ix++];          // x (32,3,300,25,2)
  const float* Amat = (const float*)d_in[ix++];          // A (3,25,25)
  const float* dbn_g = (const float*)d_in[ix++];
  const float* dbn_b = (const float*)d_in[ix++];
  const float* dbn_m = (const float*)d_in[ix++];
  const float* dbn_v = (const float*)d_in[ix++];
  struct BP {
    const float *gw,*gb,*b1g,*b1b,*b1m,*b1v,*tw,*tb,*b2g,*b2b,*b2m,*b2v;
    const float *rw,*rb,*rbg,*rbb,*rbm,*rbv,*imp;
  } bp[10];
  for (int b = 0; b < 10; ++b) {
    bp[b].gw  = (const float*)d_in[ix++];
    bp[b].gb  = (const float*)d_in[ix++];
    bp[b].b1g = (const float*)d_in[ix++];
    bp[b].b1b = (const float*)d_in[ix++];
    bp[b].b1m = (const float*)d_in[ix++];
    bp[b].b1v = (const float*)d_in[ix++];
    bp[b].tw  = (const float*)d_in[ix++];
    bp[b].tb  = (const float*)d_in[ix++];
    bp[b].b2g = (const float*)d_in[ix++];
    bp[b].b2b = (const float*)d_in[ix++];
    bp[b].b2m = (const float*)d_in[ix++];
    bp[b].b2v = (const float*)d_in[ix++];
    if (b == 4 || b == 7) {                  // only these blocks carry res params
      bp[b].rw  = (const float*)d_in[ix++];
      bp[b].rb  = (const float*)d_in[ix++];
      bp[b].rbg = (const float*)d_in[ix++];
      bp[b].rbb = (const float*)d_in[ix++];
      bp[b].rbm = (const float*)d_in[ix++];
      bp[b].rbv = (const float*)d_in[ix++];
    } else {
      bp[b].rw = bp[b].rb = bp[b].rbg = bp[b].rbb = bp[b].rbm = bp[b].rbv = nullptr;
    }
  }
  for (int b = 0; b < 10; ++b) bp[b].imp = (const float*)d_in[ix++];

  // ---- workspace carve (~492 MB) ----
  char* ws = (char*)d_ws;
  size_t off = 0;
  auto carve = [&](size_t bytes) -> void* {
    void* p = ws + off;
    off = (off + bytes + 255) & ~(size_t)255;
    return p;
  };
  __bf16* actA = (__bf16*)carve(61440000);       // activations ping
  __bf16* actB = (__bf16*)carve(61440000);       // activations pong
  __bf16* XA   = (__bf16*)carve(184320000);      // aggregated gcn input [pix][Kp]
  __bf16* H    = (__bf16*)carve(122880000);      // post-gcn hidden [pix][cout]
  __bf16* RESb = (__bf16*)carve(61440000);       // conv residual [pix][cout]
  __bf16* WG   = (__bf16*)carve((size_t)256*768*2);
  __bf16* WT   = (__bf16*)carve((size_t)256*2304*2);
  __bf16* WR   = (__bf16*)carve((size_t)256*256*2);
  float*  Ai   = (float*)carve(1875*4);
  float*  BE   = (float*)carve(256*25*4);

  const int NBm = 64, Vj = 25;

  { // data_bn -> actA (pixel-major, channel stride 3)
    long total = (long)NBm * 300 * Vj * 3;
    data_bn_kernel<<<dim3((unsigned)((total + 255) / 256)), dim3(256), 0, stream>>>(
        X, dbn_g, dbn_b, dbn_m, dbn_v, actA, total);
  }

  const int cins[10]  = {3,64,64,64,64,128,128,128,256,256};
  const int couts[10] = {64,64,64,64,128,128,128,256,256,256};
  const int strd[10]  = {1,1,1,1,2,1,1,2,1,1};

  __bf16* curIn = actA;
  int Tin = 300;
  for (int b = 0; b < 10; ++b) {
    const int cin = cins[b], cout = couts[b], st = strd[b];
    const int Tout = Tin / st;
    const int resMode = (b == 0) ? 0 : ((b == 4 || b == 7) ? 2 : 1);
    const int K3 = 3 * cin;
    const int Kp = (K3 + 31) & ~31;          // pad only block 0 (9 -> 32)
    const int Ng = NBm * Tin * Vj;           // gcn pixels
    const int Nt = NBm * Tout * Vj;          // tcn pixels
    __bf16* nextBuf = (curIn == actA) ? actB : actA;

    prep_ai_kernel<<<dim3(cdiv_i(1875,256)), dim3(256), 0, stream>>>(Amat, bp[b].imp, Ai);
    prep_bias_eff_kernel<<<dim3(cdiv_i(cout*25,256)), dim3(256), 0, stream>>>(bp[b].gb, Ai, BE, cout);
    prep_gcnw_kernel<<<dim3(cdiv_i(cout*Kp,256)), dim3(256), 0, stream>>>(bp[b].gw, WG, cout, cin, Kp);
    prep_tcnw_kernel<<<dim3(cdiv_i(cout*cout*9,256)), dim3(256), 0, stream>>>(bp[b].tw, WT, cout);
    if (resMode == 2)
      prep_resw_kernel<<<dim3(cdiv_i(cout*cin,256)), dim3(256), 0, stream>>>(bp[b].rw, WR, cout*cin);

    { // aggregation
      long total = (long)Ng * K3;
      agg_kernel<<<dim3((unsigned)((total + 255) / 256)), dim3(256), 0, stream>>>(
          curIn, Ai, XA, (long)NBm * Tin, cin, K3, Kp);
    }

    { // GCN GEMM: H = relu(bn1(WG * XA + bias_eff))
      GemmArgs ga{};
      ga.W = WG; ga.X = XA; ga.Out = H;
      ga.biasEff = BE;
      ga.bn_g = bp[b].b1g; ga.bn_b = bp[b].b1b; ga.bn_m = bp[b].b1m; ga.bn_v = bp[b].b1v;
      ga.M = cout; ga.Nn = Ng; ga.Kp = Kp; ga.V = Vj;
      gemm_kernel<0><<<dim3(cdiv_i(Ng,256), cout/32), dim3(256), 0, stream>>>(ga);
    }

    if (resMode == 2) { // residual 1x1 strided conv + bn
      GemmArgs gr{};
      gr.W = WR; gr.X = curIn; gr.Out = RESb;
      gr.cb = bp[b].rb;
      gr.bn_g = bp[b].rbg; gr.bn_b = bp[b].rbb; gr.bn_m = bp[b].rbm; gr.bn_v = bp[b].rbv;
      gr.M = cout; gr.Nn = Nt; gr.Kp = cin;
      gr.Tin = Tin; gr.Tout = Tout; gr.V = Vj; gr.stride = st; gr.cx = cin;
      gemm_kernel<2><<<dim3(cdiv_i(Nt,256), cout/32), dim3(256), 0, stream>>>(gr);
    }

    { // TCN GEMM: out = relu(bn2(WT * im2col(H) + b) + res)
      GemmArgs gt{};
      gt.W = WT; gt.X = H; gt.Out = nextBuf;
      gt.cb = bp[b].tb;
      gt.bn_g = bp[b].b2g; gt.bn_b = bp[b].b2b; gt.bn_m = bp[b].b2m; gt.bn_v = bp[b].b2v;
      gt.Res = (resMode == 2) ? RESb : curIn;
      gt.OutF = (b == 9) ? (float*)d_out : nullptr;
      gt.M = cout; gt.Nn = Nt; gt.Kp = cout * 9;
      gt.Tin = Tin; gt.Tout = Tout; gt.V = Vj; gt.stride = st; gt.cx = cout;
      gt.resMode = resMode;
      gemm_kernel<1><<<dim3(cdiv_i(Nt,256), cout/32), dim3(256), 0, stream>>>(gt);
    }

    curIn = nextBuf;
    Tin = Tout;
  }
}
